// QwenAttention_61761629716694
// MI455X (gfx1250) — matrix operand order repaired and checked by reading, not compiled, not run on hardware
//
#include <hip/hip_runtime.h>
#include <hip/hip_bf16.h>
#include <math.h>

typedef __bf16 bf16;
typedef __attribute__((ext_vector_type(16))) __bf16 v16bf;
typedef __attribute__((ext_vector_type(8)))  __bf16 v8bf;
typedef __attribute__((ext_vector_type(8)))  float  v8f;
typedef __attribute__((ext_vector_type(4)))  unsigned int v4u;
typedef __attribute__((ext_vector_type(8)))  int v8i;
typedef __attribute__((ext_vector_type(4)))  int v4i;

#if defined(__has_builtin)
#  if __has_builtin(__builtin_amdgcn_tensor_load_to_lds) && \
      __has_builtin(__builtin_amdgcn_s_wait_tensorcnt)
#    define HAVE_TDM 1
#  endif
#endif
#ifndef HAVE_TDM
#define HAVE_TDM 0
#endif

constexpr int B_  = 2;
constexpr int T_  = 2048;
constexpr int C_  = 2048;
constexpr int H_  = 16;
constexpr int HKV_ = 4;
constexpr int DH_ = 128;
constexpr int GROUPS_ = H_ / HKV_;
constexpr int MROWS = B_ * T_;            // 4096
constexpr float SCALE = 0.0883883476483184f; // 1/sqrt(128), logn==1 since T==MTL

// LDS staging geometry for the GEMM B-panel (64 rows x 256-element K chunks)
constexpr int KCH = 256;                   // bf16 elements per K chunk per row
constexpr int ROW_ELEMS = 264;             // 256 data + 8 pad elems (16B TDM pad)
constexpr int BUF_ELEMS = 64 * ROW_ELEMS;  // 16896 elems = 33792 B per buffer
constexpr int BUF_BYTES = BUF_ELEMS * 2;

// ---------- WMMA helpers ----------

__device__ __forceinline__ v8f wmma_bf16(v16bf a, v16bf b, v8f c) {
  // (neg_a, A, neg_b, B, c_mod, C, reuse_a, reuse_b)
  return __builtin_amdgcn_wmma_f32_16x16x32_bf16(false, a, false, b, (short)0, c,
                                                 false, false);
}

// A operand: 16x32 tile of row-major [rows x ld] at (row0, k0).
// lane L: row row0+(L&15); elems 0..15 = K kk..kk+15,
// kk = k0 + ((L&16)?16:0).
__device__ __forceinline__ v16bf load_a_frag(const bf16* base, int ld, int row0,
                                             int k0, int lane) {
  const bf16* p = base + (size_t)(row0 + (lane & 15)) * ld
                       + (k0 + ((lane & 16) ? 16 : 0));
  v8bf lo = *(const v8bf*)p;
  v8bf hi = *(const v8bf*)(p + 8);
  return __builtin_shufflevector(lo, hi, 0,1,2,3,4,5,6,7,8,9,10,11,12,13,14,15);
}

// B operand from global for D = A * Wrows^T: lane L = column col0+(L&15),
// elems 0..15 = K k0..k0+15 (lanes 0-15) or k0+16..k0+31 (lanes 16-31).
__device__ __forceinline__ v16bf load_b_frag(const bf16* base, int ld, int col0,
                                             int k0, int lane) {
  const bf16* p = base + (size_t)(col0 + (lane & 15)) * ld
                       + (k0 + ((lane & 16) ? 16 : 0));
  return *(const v16bf*)p;
}

// 16 contiguous bf16 as two 16B loads (ds_load_b128-friendly; 16B alignment ok)
__device__ __forceinline__ v16bf load16_contig(const bf16* p) {
  v8bf lo = *(const v8bf*)p;
  v8bf hi = *(const v8bf*)(p + 8);
  return __builtin_shufflevector(lo, hi, 0,1,2,3,4,5,6,7,8,9,10,11,12,13,14,15);
}

// ---------- TDM: load one 64 x 256-elem bf16 tile (row stride K) into LDS ----------
// Padded: pad_enable, interval=128 DW (one row = 512B), amount=4 DW (16B)
// -> LDS row stride 528B, conflict-free 16-row fragment reads.

__device__ __forceinline__ void tdm_load_b_chunk(const bf16* gsrc,
                                                 unsigned lds_off, int K) {
#if HAVE_TDM
  unsigned long long ga = (unsigned long long)gsrc;
  v4u g0 = { 1u,                                  // count=1
             lds_off,                             // lds_addr
             (unsigned)(ga & 0xFFFFFFFFu),        // global_addr[31:0]
             (unsigned)((ga >> 32) & 0x01FFFFFFu) | (2u << 30) }; // ga hi | type=2
  v8i g1 = { (int)((1u << 16)       // data_size = 1 (2 bytes)
                 | (1u << 20)       // pad_enable
                 | (6u << 22)       // pad_interval: 128 DWORDs
                 | (3u << 25)),     // pad_amount: 4 DWORDs (16B)
             (int)((unsigned)KCH << 16),   // tensor_dim0 = 256 (lo16 in [63:48])
             (int)(64u << 16),             // tensor_dim0 hi=0 | tensor_dim1 = 64
             (int)((unsigned)KCH << 16),   // tensor_dim1 hi=0 | tile_dim0 = 256
             64,                           // tile_dim1 = 64, tile_dim2 = 0
             K,                            // tensor_dim0_stride = K (row stride)
             0, 0 };
  v4i g2 = { 0, 0, 0, 0 };
  v4i g3 = { 0, 0, 0, 0 };
#if __clang_major__ >= 23
  v8i g4 = { 0, 0, 0, 0, 0, 0, 0, 0 };
  __builtin_amdgcn_tensor_load_to_lds(g0, g1, g2, g3, g4, 0);
#else
  __builtin_amdgcn_tensor_load_to_lds(g0, g1, g2, g3, 0);
#endif
#else
  (void)gsrc; (void)lds_off; (void)K;
#endif
}

// ---------- elementwise kernels ----------

__global__ void cvt_f32_bf16(const float* __restrict__ src, bf16* __restrict__ dst,
                             int n) {
  int i = blockIdx.x * blockDim.x + threadIdx.x;
  if (i < n) dst[i] = (bf16)src[i];
}

// RoPE + repack [B*T, nh*DH] f32 -> [B, nh, T, DH] bf16
__global__ void rope_pack(const float* __restrict__ tmp, const float* __restrict__ cs,
                          const float* __restrict__ sn, bf16* __restrict__ outp,
                          int nh) {
  int tid = blockIdx.x * blockDim.x + threadIdx.x;
  int total = B_ * T_ * nh * (DH_ / 2);
  if (tid >= total) return;
  int p = tid & 63;
  int h = (tid >> 6) % nh;
  int t = (tid / (64 * nh)) % T_;
  int b = tid / (64 * nh * T_);
  const float* row = tmp + (size_t)(b * T_ + t) * (nh * DH_) + h * DH_ + 2 * p;
  float x0 = row[0], x1 = row[1];
  float c = cs[t * 64 + p], s = sn[t * 64 + p];
  bf16* o = outp + ((size_t)((b * nh + h) * T_ + t)) * DH_ + 2 * p;
  o[0] = (bf16)(x0 * c - x1 * s);
  o[1] = (bf16)(x0 * s + x1 * c);
}

// V: [B*T, HKV*DH] f32 -> [B, HKV, DH, T] bf16 (PV B-operand contiguity)
__global__ void vpack_t(const float* __restrict__ tmp, bf16* __restrict__ outp) {
  int tid = blockIdx.x * blockDim.x + threadIdx.x;
  int total = B_ * HKV_ * T_ * DH_;
  if (tid >= total) return;
  int d = tid % DH_;
  int t = (tid / DH_) % T_;
  int h = (tid / (DH_ * T_)) % HKV_;
  int b = tid / (DH_ * T_ * HKV_);
  float v = tmp[(size_t)(b * T_ + t) * (HKV_ * DH_) + h * DH_ + d];
  outp[((size_t)((b * HKV_ + h) * DH_ + d)) * T_ + t] = (bf16)v;
}

// ---------- GEMM: out[M,N] = A[M,K](bf16) * W[N,K](bf16)^T + bias ----------
// Block = 8 waves = 128 rows x 64 cols. B panel (64 x K) staged through LDS in
// 256-element K chunks, double-buffered via TDM (tensor_load_to_lds +
// s_wait_tensorcnt). A fragments stream from global (L2-resident).

__global__ __launch_bounds__(256)
void gemm_xwT_lds(const bf16* __restrict__ A, const bf16* __restrict__ W,
                  const float* __restrict__ bias, float* __restrict__ out,
                  int M, int N, int K) {
  extern __shared__ bf16 ldsB[];   // 2 * BUF_ELEMS
  int lane = threadIdx.x & 31;
  int wv   = threadIdx.x >> 5;
  int mblocks = M >> 7;
  int bm = blockIdx.x % mblocks;
  int bn = blockIdx.x / mblocks;
  int row0 = (bm << 7) + (wv << 4);
  int col0 = bn << 6;

  v8f acc0 = {}, acc1 = {}, acc2 = {}, acc3 = {};
  int nch = K / KCH;
  const bf16* wpanel = W + (size_t)col0 * K;

#if HAVE_TDM
  unsigned lds_base = (unsigned)(unsigned long long)(void*)ldsB;
  if (wv == 0) tdm_load_b_chunk(wpanel, lds_base, K);
  for (int ch = 0; ch < nch; ++ch) {
    if (wv == 0) {
      if (ch + 1 < nch) {
        tdm_load_b_chunk(wpanel + (ch + 1) * KCH,
                         lds_base + (unsigned)(((ch + 1) & 1) * BUF_BYTES), K);
        __builtin_amdgcn_s_wait_tensorcnt(1);   // chunk ch complete (in-order)
      } else {
        __builtin_amdgcn_s_wait_tensorcnt(0);
      }
    }
    __syncthreads();
    const bf16* bb = ldsB + (ch & 1) * BUF_ELEMS;
#pragma unroll
    for (int ks = 0; ks < KCH / 32; ++ks) {
      int koff = ks * 32 + ((lane & 16) ? 16 : 0);
      v16bf a  = load_a_frag(A, K, row0, ch * KCH + ks * 32, lane);
      v16bf b0 = load16_contig(bb + ((lane & 15) +  0) * ROW_ELEMS + koff);
      v16bf b1 = load16_contig(bb + ((lane & 15) + 16) * ROW_ELEMS + koff);
      v16bf b2 = load16_contig(bb + ((lane & 15) + 32) * ROW_ELEMS + koff);
      v16bf b3 = load16_contig(bb + ((lane & 15) + 48) * ROW_ELEMS + koff);
      acc0 = wmma_bf16(a, b0, acc0);
      acc1 = wmma_bf16(a, b1, acc1);
      acc2 = wmma_bf16(a, b2, acc2);
      acc3 = wmma_bf16(a, b3, acc3);
    }
    __syncthreads();
  }
#else
  // fallback: cooperative copy into the same padded layout, single buffer
  for (int ch = 0; ch < nch; ++ch) {
    __syncthreads();
    for (int idx = threadIdx.x; idx < 64 * 32; idx += 256) {
      int r = idx >> 5, c = idx & 31;      // 32 x 16B per 512B row
      const char* s = (const char*)(wpanel + (size_t)r * K + ch * KCH) + c * 16;
      char* d = (char*)ldsB + r * (ROW_ELEMS * 2) + c * 16;
      *(v8bf*)d = *(const v8bf*)s;
    }
    __syncthreads();
    const bf16* bb = ldsB;
#pragma unroll
    for (int ks = 0; ks < KCH / 32; ++ks) {
      int koff = ks * 32 + ((lane & 16) ? 16 : 0);
      v16bf a  = load_a_frag(A, K, row0, ch * KCH + ks * 32, lane);
      v16bf b0 = load16_contig(bb + ((lane & 15) +  0) * ROW_ELEMS + koff);
      v16bf b1 = load16_contig(bb + ((lane & 15) + 16) * ROW_ELEMS + koff);
      v16bf b2 = load16_contig(bb + ((lane & 15) + 32) * ROW_ELEMS + koff);
      v16bf b3 = load16_contig(bb + ((lane & 15) + 48) * ROW_ELEMS + koff);
      acc0 = wmma_bf16(a, b0, acc0);
      acc1 = wmma_bf16(a, b1, acc1);
      acc2 = wmma_bf16(a, b2, acc2);
      acc3 = wmma_bf16(a, b3, acc3);
    }
  }
#endif

  int rsel = (lane & 16) ? 8 : 0;
  v8f acc[4] = {acc0, acc1, acc2, acc3};
#pragma unroll
  for (int j = 0; j < 4; ++j) {
    int col = col0 + j * 16 + (lane & 15);
    float bv = bias ? bias[col] : 0.0f;
#pragma unroll
    for (int i = 0; i < 8; ++i) {
      int row = row0 + rsel + i;
      out[(size_t)row * N + col] = acc[j][i] + bv;
    }
  }
}

// ---------- flash attention ----------
// q: [B,H,T,DH] bf16 (RoPE'd)  k: [B,HKV,T,DH] bf16 (RoPE'd)
// vt: [B,HKV,DH,T] bf16        attn: [B*T, C] bf16
// one wave per 16-query tile; online softmax in WMMA C-layout.

__global__ __launch_bounds__(128)
void flash_attn(const bf16* __restrict__ q, const bf16* __restrict__ k,
                const bf16* __restrict__ vt, bf16* __restrict__ attn) {
  __shared__ bf16 lds_p[4][16 * 32];

  int lane = threadIdx.x & 31;
  int wv   = threadIdx.x >> 5;
  int wid  = blockIdx.x * 4 + wv;
  int qtiles = T_ / 16;
  int total  = B_ * H_ * qtiles;
  if (wid >= total) return;

  int qt = wid % qtiles;
  int h  = (wid / qtiles) % H_;
  int b  = wid / (qtiles * H_);
  int kvh = h / GROUPS_;
  int qbase = qt * 16;

  const bf16* qptr = q  + ((size_t)(b * H_   + h  ) * T_) * DH_;
  const bf16* kptr = k  + ((size_t)(b * HKV_ + kvh) * T_) * DH_;
  const bf16* vptr = vt + ((size_t)(b * HKV_ + kvh) * DH_) * T_;
  bf16* lp = &lds_p[wv][0];

  v16bf qa[4];
#pragma unroll
  for (int kc = 0; kc < 4; ++kc)
    qa[kc] = load_a_frag(qptr, DH_, qbase, kc * 32, lane);

  float m[8], l[8];
  v8f o[8];
#pragma unroll
  for (int i = 0; i < 8; ++i) { m[i] = -1e30f; l[i] = 0.f; o[i] = (v8f){}; }

  int rsel = (lane & 16) ? 8 : 0;
  int kend = qbase + 16;

  for (int kb = 0; kb < kend; kb += 32) {
    v8f s0 = {}, s1 = {};
#pragma unroll
    for (int kc = 0; kc < 4; ++kc) {
      v16bf kb0 = load_b_frag(kptr, DH_, kb,      kc * 32, lane);
      v16bf kb1 = load_b_frag(kptr, DH_, kb + 16, kc * 32, lane);
      s0 = wmma_bf16(qa[kc], kb0, s0);
      s1 = wmma_bf16(qa[kc], kb1, s1);
    }

    int c0 = kb + (lane & 15);
    int c1 = c0 + 16;
    float sf[8];
#pragma unroll
    for (int i = 0; i < 8; ++i) {
      int row = qbase + rsel + i;
      bool ok0 = (c0 <= row), ok1 = (c1 <= row);
      float v0 = ok0 ? s0[i] * SCALE : -3.0e38f;
      float v1 = ok1 ? s1[i] * SCALE : -3.0e38f;
      float t = fmaxf(v0, v1);
      t = fmaxf(t, __shfl_xor(t, 1, 32));
      t = fmaxf(t, __shfl_xor(t, 2, 32));
      t = fmaxf(t, __shfl_xor(t, 4, 32));
      t = fmaxf(t, __shfl_xor(t, 8, 32));
      float mn = fmaxf(m[i], t);
      float p0 = ok0 ? __expf(v0 - mn) : 0.f;
      float p1 = ok1 ? __expf(v1 - mn) : 0.f;
      float rs = p0 + p1;
      rs += __shfl_xor(rs, 1, 32);
      rs += __shfl_xor(rs, 2, 32);
      rs += __shfl_xor(rs, 4, 32);
      rs += __shfl_xor(rs, 8, 32);
      sf[i] = __expf(m[i] - mn);
      m[i] = mn;
      l[i] = l[i] * sf[i] + rs;
      int lrow = i + rsel;
      lp[lrow * 32 +      (lane & 15)] = (bf16)p0;
      lp[lrow * 32 + 16 + (lane & 15)] = (bf16)p1;
    }
#pragma unroll
    for (int cc = 0; cc < 8; ++cc)
#pragma unroll
      for (int i = 0; i < 8; ++i) o[cc][i] = o[cc][i] * sf[i];

    asm volatile("s_wait_dscnt 0" ::: "memory");

    const bf16* pp = lp + (lane & 15) * 32 + ((lane & 16) ? 16 : 0);
    v8bf plo = *(const v8bf*)pp;
    v8bf phi = *(const v8bf*)(pp + 8);
    v16bf pa = __builtin_shufflevector(plo, phi,
                 0,1,2,3,4,5,6,7,8,9,10,11,12,13,14,15);

#pragma unroll
    for (int cc = 0; cc < 8; ++cc) {
      v16bf vb = load_b_frag(vptr, T_, cc * 16, kb, lane);
      o[cc] = wmma_bf16(pa, vb, o[cc]);
    }
  }

#pragma unroll
  for (int cc = 0; cc < 8; ++cc) {
#pragma unroll
    for (int i = 0; i < 8; ++i) {
      int row = qbase + rsel + i;
      float val = o[cc][i] / l[i];
      attn[((size_t)(b * T_ + row)) * C_ + h * DH_ + cc * 16 + (lane & 15)] =
          (bf16)val;
    }
  }
}

// ---------- launch ----------

static inline size_t align256(size_t x) { return (x + 255) & ~(size_t)255; }

extern "C" void kernel_launch(void* const* d_in, const int* in_sizes, int n_in,
                              void* d_out, int out_size, void* d_ws, size_t ws_size,
                              hipStream_t stream) {
  (void)in_sizes; (void)n_in; (void)out_size; (void)ws_size;
  const float* x   = (const float*)d_in[0];
  const float* Wq  = (const float*)d_in[1];
  const float* bq  = (const float*)d_in[2];
  const float* Wk  = (const float*)d_in[3];
  const float* bk  = (const float*)d_in[4];
  const float* Wv  = (const float*)d_in[5];
  const float* bv  = (const float*)d_in[6];
  const float* Wo  = (const float*)d_in[7];
  const float* cs  = (const float*)d_in[8];
  const float* sn  = (const float*)d_in[9];
  float* out = (float*)d_out;

  const int NQ = H_ * DH_;     // 2048
  const int NKV = HKV_ * DH_;  // 512

  char* w = (char*)d_ws;
  size_t off = 0;
  auto carve = [&](size_t bytes) { void* p = w + off; off += align256(bytes); return p; };
  bf16* xb   = (bf16*)carve((size_t)MROWS * C_ * 2);
  bf16* wqb  = (bf16*)carve((size_t)NQ * C_ * 2);
  bf16* wkb  = (bf16*)carve((size_t)NKV * C_ * 2);
  bf16* wvb  = (bf16*)carve((size_t)NKV * C_ * 2);
  bf16* wob  = (bf16*)carve((size_t)C_ * C_ * 2);
  float* qtmp = (float*)carve((size_t)MROWS * NQ * 4);
  float* ktmp = (float*)carve((size_t)MROWS * NKV * 4);
  float* vtmp = (float*)carve((size_t)MROWS * NKV * 4);
  bf16* qr   = (bf16*)carve((size_t)B_ * H_ * T_ * DH_ * 2);
  bf16* kr   = (bf16*)carve((size_t)B_ * HKV_ * T_ * DH_ * 2);
  bf16* vtb  = (bf16*)carve((size_t)B_ * HKV_ * DH_ * T_ * 2);
  bf16* attn = (bf16*)carve((size_t)MROWS * C_ * 2);

  auto cvt = [&](const float* s, bf16* d, int n) {
    cvt_f32_bf16<<<(n + 255) / 256, 256, 0, stream>>>(s, d, n);
  };
  cvt(x,  xb,  MROWS * C_);
  cvt(Wq, wqb, NQ * C_);
  cvt(Wk, wkb, NKV * C_);
  cvt(Wv, wvb, NKV * C_);
  cvt(Wo, wob, C_ * C_);

  auto gemm = [&](const bf16* A, const bf16* W_, const float* bias, float* O,
                  int M, int N, int K) {
    int blocks = (M / 128) * (N / 64);
    gemm_xwT_lds<<<blocks, 256, 2 * BUF_BYTES, stream>>>(A, W_, bias, O, M, N, K);
  };
  gemm(xb, wqb, bq, qtmp, MROWS, NQ, C_);
  gemm(xb, wkb, bk, ktmp, MROWS, NKV, C_);
  gemm(xb, wvb, bv, vtmp, MROWS, NKV, C_);

  {
    int n = B_ * T_ * H_ * (DH_ / 2);
    rope_pack<<<(n + 255) / 256, 256, 0, stream>>>(qtmp, cs, sn, qr, H_);
  }
  {
    int n = B_ * T_ * HKV_ * (DH_ / 2);
    rope_pack<<<(n + 255) / 256, 256, 0, stream>>>(ktmp, cs, sn, kr, HKV_);
  }
  {
    int n = B_ * HKV_ * T_ * DH_;
    vpack_t<<<(n + 255) / 256, 256, 0, stream>>>(vtmp, vtb);
  }

  {
    int tiles = B_ * H_ * (T_ / 16);
    flash_attn<<<tiles / 4, 128, 0, stream>>>(qr, kr, vtb, attn);
  }

  gemm(attn, wob, nullptr, out, MROWS, C_, C_);
}
